// MultiHeadAttention_72121090834917
// MI455X (gfx1250) — matrix operand order repaired and checked by reading, not compiled, not run on hardware
//
#include <hip/hip_runtime.h>
#include <hip/hip_bf16.h>

// ---------------------------------------------------------------------------
// MHA forward, fused flash-attention, bf16 WMMA (v_wmma_f32_16x16x32_bf16)
// B=2, S=2048, H=1024, NH=16, HD=64
// Round 4: fix the async-to-LDS builtin call signature — clang reported the
// expected param type as v4i in addrspace(1) (global src) / addrspace(3)
// (LDS dst), non-const. Attention K/V staging stays double-buffered with
// GLOBAL_LOAD_ASYNC_TO_LDS_B128 overlapping WMMA compute; GEMMs keep the
// 64x32 register-blocked shape (12 B128 loads -> 8 WMMAs / K-step).
// ---------------------------------------------------------------------------

#define B_   2
#define S_   2048
#define H_   1024
#define NH_  16
#define HD_  64
#define M_   (B_ * S_)          // 4096 rows of X

typedef __attribute__((ext_vector_type(16))) __bf16 v16bf;
typedef __attribute__((ext_vector_type(8)))  float  v8f;
typedef int v4i __attribute__((vector_size(16)));

union FragU { v16bf v; uint4 q[2]; };

#if defined(__has_builtin)
#  if __has_builtin(__builtin_amdgcn_global_load_async_to_lds_b128)
#    define USE_ASYNC_LDS 1
#  endif
#endif
#ifndef USE_ASYNC_LDS
#  define USE_ASYNC_LDS 0
#endif

// 16-byte global -> LDS copy. Async (ASYNCcnt-tracked, no VGPR round trip)
// when the gfx1250 builtin exists; plain load+ds_store otherwise.
__device__ __forceinline__ void copy16_to_lds(uint4* lds_dst, const uint4* gsrc) {
#if USE_ASYNC_LDS
    __builtin_amdgcn_global_load_async_to_lds_b128(
        (__attribute__((address_space(1))) v4i*)gsrc,
        (__attribute__((address_space(3))) v4i*)lds_dst, 0, 0);
#else
    *lds_dst = *gsrc;
#endif
}
__device__ __forceinline__ void copy_wait() {
#if USE_ASYNC_LDS
    asm volatile("s_wait_asynccnt 0" ::: "memory");
#endif
}

// A-matrix fragment, 16x32 bf16, source row-major [16 x stride], rows = lane&15.
// ISA layout: lanes 0-15 -> VGPR0-3: K=0..7, VGPR4-7: K=16..23; lanes 16-31: +8.
__device__ __forceinline__ v16bf load_frag_a(const __bf16* base, int stride, int kb, int lane) {
    FragU f;
    const __bf16* p = base + (lane & 15) * stride + kb + ((lane >> 4) & 1) * 8;
    f.q[0] = *(const uint4*)(p);        // K = off+0..7
    f.q[1] = *(const uint4*)(p + 16);   // K = off+16..23
    return f.v;
}

// B-matrix fragment, 32x16 bf16. Element (k,n) at base[n*stride + k]
// (source holds columns of B contiguously in k — exactly a weight row).
// ISA layout: lanes 0-15 -> VGPR0-3: K=0..7, VGPR4-7: K=16..23; lanes 16-31: +8.
__device__ __forceinline__ v16bf load_frag_b(const __bf16* base, int stride, int kb, int lane) {
    FragU f;
    const __bf16* p = base + (lane & 15) * stride + kb + ((lane >> 4) & 1) * 8;
    f.q[0] = *(const uint4*)(p);        // K = off+0..7
    f.q[1] = *(const uint4*)(p + 16);   // K = off+16..23
    return f.v;
}

__device__ __forceinline__ v8f wmma_bf16(v16bf a, v16bf b, v8f c) {
    return __builtin_amdgcn_wmma_f32_16x16x32_bf16(false, a, false, b, (short)0, c, false, false);
}

__device__ __forceinline__ float half16_max(float v) {
    v = fmaxf(v, __shfl_xor(v, 1, 32));
    v = fmaxf(v, __shfl_xor(v, 2, 32));
    v = fmaxf(v, __shfl_xor(v, 4, 32));
    v = fmaxf(v, __shfl_xor(v, 8, 32));
    return v;
}
__device__ __forceinline__ float half16_sum(float v) {
    v += __shfl_xor(v, 1, 32);
    v += __shfl_xor(v, 2, 32);
    v += __shfl_xor(v, 4, 32);
    v += __shfl_xor(v, 8, 32);
    return v;
}

// ---------------------------------------------------------------------------
// Stage 0: fp32 -> bf16 conversion (4-wide)
// ---------------------------------------------------------------------------
__global__ void f2bf4_kernel(const float* __restrict__ in, __bf16* __restrict__ out, int n4) {
    int i = blockIdx.x * blockDim.x + threadIdx.x;
    if (i < n4) {
        float4 f = ((const float4*)in)[i];
        out[4 * i + 0] = (__bf16)f.x;
        out[4 * i + 1] = (__bf16)f.y;
        out[4 * i + 2] = (__bf16)f.z;
        out[4 * i + 3] = (__bf16)f.w;
    }
}

// ---------------------------------------------------------------------------
// Stage 1: fused QKV projection. One wave per 64x32 macro-tile (4 M x 2 N).
// Per K-step: 4 A-frags + 2 B-frags -> 8 WMMAs (4x/2x register reuse).
// N columns [0,1024)->Q (scaled), [1024,2048)->K, [2048,3072)->V (transposed).
// ---------------------------------------------------------------------------
__global__ void qkv_gemm_kernel(const __bf16* __restrict__ Xb,
                                const __bf16* __restrict__ Wqb,
                                const __bf16* __restrict__ Wkb,
                                const __bf16* __restrict__ Wvb,
                                const float* __restrict__ bq,
                                const float* __restrict__ bk,
                                const float* __restrict__ bv,
                                __bf16* __restrict__ Qh,
                                __bf16* __restrict__ Kh,
                                __bf16* __restrict__ Vt) {
    const int lane = threadIdx.x & 31;
    const int tile = blockIdx.x * (blockDim.x >> 5) + (threadIdx.x >> 5);
    const int ng = tile % 96;               // 96 N-groups of 32 cols (3072 total)
    const int mg = tile / 96;               // 64 M-groups of 64 rows
    const int n0 = ng * 32;                 // never straddles a 1024 boundary
    const int which = n0 >> 10;             // 0:Q 1:K 2:V
    const int nn0 = n0 & 1023;              // column within selected matrix

    const __bf16* W  = (which == 0) ? Wqb : (which == 1) ? Wkb : Wvb;
    const float*  bi = (which == 0) ? bq  : (which == 1) ? bk  : bv;

    v8f acc[4][2];
#pragma unroll
    for (int i = 0; i < 4; ++i)
#pragma unroll
        for (int j = 0; j < 2; ++j) acc[i][j] = v8f{0.f,0.f,0.f,0.f,0.f,0.f,0.f,0.f};

    const __bf16* Abase = Xb + (size_t)mg * 64 * H_;
    const __bf16* Bbase = W + (size_t)nn0 * H_;
    for (int kb = 0; kb < H_; kb += 32) {
        v16bf a0 = load_frag_a(Abase + 0 * 16 * H_, H_, kb, lane);
        v16bf a1 = load_frag_a(Abase + 1 * 16 * H_, H_, kb, lane);
        v16bf a2 = load_frag_a(Abase + 2 * 16 * H_, H_, kb, lane);
        v16bf a3 = load_frag_a(Abase + 3 * 16 * H_, H_, kb, lane);
        v16bf b0 = load_frag_b(Bbase + 0 * 16 * H_, H_, kb, lane);
        v16bf b1 = load_frag_b(Bbase + 1 * 16 * H_, H_, kb, lane);
        acc[0][0] = wmma_bf16(a0, b0, acc[0][0]);
        acc[0][1] = wmma_bf16(a0, b1, acc[0][1]);
        acc[1][0] = wmma_bf16(a1, b0, acc[1][0]);
        acc[1][1] = wmma_bf16(a1, b1, acc[1][1]);
        acc[2][0] = wmma_bf16(a2, b0, acc[2][0]);
        acc[2][1] = wmma_bf16(a2, b1, acc[2][1]);
        acc[3][0] = wmma_bf16(a3, b0, acc[3][0]);
        acc[3][1] = wmma_bf16(a3, b1, acc[3][1]);
    }

    const int rowbase = ((lane >> 4) & 1) * 8;
#pragma unroll
    for (int j = 0; j < 2; ++j) {
        const int n = nn0 + j * 16 + (lane & 15);
        const int h = n >> 6, d = n & 63;
        const float bias = bi[n];
#pragma unroll
        for (int i = 0; i < 4; ++i) {
#pragma unroll
            for (int r = 0; r < 8; ++r) {
                int m = mg * 64 + i * 16 + rowbase + r;
                int bidx = m >> 11, s = m & (S_ - 1);
                float v = acc[i][j][r] + bias;
                if (which == 0) {
                    v *= 0.125f;  // 1/sqrt(64) folded into Q
                    Qh[((bidx * NH_ + h) * S_ + s) * HD_ + d] = (__bf16)v;
                } else if (which == 1) {
                    Kh[((bidx * NH_ + h) * S_ + s) * HD_ + d] = (__bf16)v;
                } else {
                    Vt[((bidx * NH_ + h) * HD_ + d) * S_ + s] = (__bf16)v;
                }
            }
        }
    }
}

// ---------------------------------------------------------------------------
// Stage 2: fused causal flash attention.
// Block = 8 waves sharing one (b,h) and 8 consecutive q-tiles (128 q rows).
// Per 64-key step the block stages K (64x64) and V^T (64x64) into LDS once;
// tiles are double-buffered and filled by async global->LDS copies so the
// staging of step st+1 overlaps the WMMA compute of step st.
// ---------------------------------------------------------------------------
__device__ __forceinline__ void stage_kv(int tid,
                                         const __bf16* __restrict__ Kp,
                                         const __bf16* __restrict__ Vp,
                                         int k0,
                                         __bf16* __restrict__ Kbuf,
                                         __bf16* __restrict__ Vbuf) {
    const uint4* Kg = (const uint4*)(Kp + (size_t)k0 * HD_);  // K tile is contiguous
    uint4* Ks = (uint4*)Kbuf;
    uint4* Vs = (uint4*)Vbuf;
#pragma unroll
    for (int it = 0; it < 2; ++it) {
        int i = tid + it * 256;                         // 512 uint4 per tile
        copy16_to_lds(&Ks[i], &Kg[i]);
        int d = i >> 3, c = i & 7;                      // 8 uint4 per 64-key V row
        copy16_to_lds(&Vs[i], (const uint4*)(Vp + (size_t)d * S_ + k0 + c * 8));
    }
}

__global__ void attn_kernel(const __bf16* __restrict__ Q,
                            const __bf16* __restrict__ K,
                            const __bf16* __restrict__ Vt,
                            __bf16* __restrict__ Ctx) {
    __shared__ alignas(16) __bf16 Klds[2][64 * 64];   // [buf][key_rel][d]
    __shared__ alignas(16) __bf16 Vlds[2][64 * 64];   // [buf][d][key_rel]
    __shared__ alignas(16) __bf16 plds[8][16 * 32];   // per-wave P staging (C->A layout)

    const int tid = threadIdx.x;
    const int lane = tid & 31;
    const int wave = tid >> 5;
    const int bh = blockIdx.x >> 4;             // (b*NH + h) in [0,32)
    const int qblock = blockIdx.x & 15;         // 16 q-blocks of 128 rows
    const int qt = qblock * 8 + wave;           // this wave's 16-row q tile

    const __bf16* Qp = Q + (size_t)bh * S_ * HD_;
    const __bf16* Kp = K + (size_t)bh * S_ * HD_;
    const __bf16* Vp = Vt + (size_t)bh * HD_ * S_;

    const v16bf qf0 = load_frag_a(Qp + qt * 16 * HD_, HD_, 0, lane);
    const v16bf qf1 = load_frag_a(Qp + qt * 16 * HD_, HD_, 32, lane);

    const float NEG_INF = -__builtin_inff();
    float mrow[8], lrow[8];
    v8f o[4];
#pragma unroll
    for (int r = 0; r < 8; ++r) { mrow[r] = NEG_INF; lrow[r] = 0.f; }
#pragma unroll
    for (int f = 0; f < 4; ++f) o[f] = v8f{0.f,0.f,0.f,0.f,0.f,0.f,0.f,0.f};

    const int rowbase = ((lane >> 4) & 1) * 8;
    const int ncol = lane & 15;
    const int qg0 = qt * 16 + rowbase;
    const int q_hi = qt * 16 + 15;
    const int nsteps = (qblock + 1) * 2;        // 64-key steps; block-uniform

    // prologue: stage tile 0 into buffer 0
    stage_kv(tid, Kp, Vp, 0, Klds[0], Vlds[0]);
    copy_wait();
    __syncthreads();

    for (int st = 0; st < nsteps; ++st) {
        const int k0 = st * 64;
        const int cur = st & 1;

        // kick off async staging of the next tile into the other buffer
        if (st + 1 < nsteps)
            stage_kv(tid, Kp, Vp, (st + 1) * 64, Klds[cur ^ 1], Vlds[cur ^ 1]);

        // ---- per-wave compute on the current buffer (two 32-key substeps) ----
        const __bf16* Kb = Klds[cur];
        const __bf16* Vb = Vlds[cur];
#pragma unroll
        for (int sub = 0; sub < 2; ++sub) {
            const int ks = k0 + sub * 32;       // global key base of this substep
            if (ks > q_hi) continue;            // entirely beyond causal range

            v8f s0 = {0.f,0.f,0.f,0.f,0.f,0.f,0.f,0.f};
            v8f s1 = {0.f,0.f,0.f,0.f,0.f,0.f,0.f,0.f};
            const __bf16* Kt = Kb + (sub * 32) * 64;
            s0 = wmma_bf16(qf0, load_frag_b(Kt + 0 * 64,  64, 0,  lane), s0);
            s0 = wmma_bf16(qf1, load_frag_b(Kt + 0 * 64,  64, 32, lane), s0);
            s1 = wmma_bf16(qf0, load_frag_b(Kt + 16 * 64, 64, 0,  lane), s1);
            s1 = wmma_bf16(qf1, load_frag_b(Kt + 16 * 64, 64, 32, lane), s1);

            float alpha[8];
#pragma unroll
            for (int r = 0; r < 8; ++r) {
                const int qg = qg0 + r;
                float a0 = (ks + ncol > qg) ? NEG_INF : s0[r];
                float a1 = (ks + 16 + ncol > qg) ? NEG_INF : s1[r];
                float mn = fmaxf(mrow[r], half16_max(fmaxf(a0, a1)));
                float p0 = __expf(a0 - mn);
                float p1 = __expf(a1 - mn);
                float rs = half16_sum(p0 + p1);
                float al = __expf(mrow[r] - mn);
                lrow[r] = lrow[r] * al + rs;
                mrow[r] = mn;
                alpha[r] = al;
                s0[r] = p0;
                s1[r] = p1;
            }
#pragma unroll
            for (int f = 0; f < 4; ++f)
#pragma unroll
                for (int r = 0; r < 8; ++r) o[f][r] *= alpha[r];

            // P: C-layout -> A-layout via per-wave LDS tile (16x32, stride 32)
            __bf16* pw = plds[wave];
#pragma unroll
            for (int r = 0; r < 8; ++r) {
                pw[(rowbase + r) * 32 + ncol] = (__bf16)s0[r];
                pw[(rowbase + r) * 32 + 16 + ncol] = (__bf16)s1[r];
            }
            asm volatile("s_wait_dscnt 0" ::: "memory");
            v16bf pf = load_frag_a(pw, 32, 0, lane);

            // O += P @ V : element (k=key_rel, n=d) at Vb[(f*16+n)*64 + sub*32 + k]
#pragma unroll
            for (int f = 0; f < 4; ++f) {
                v16bf vf = load_frag_b(Vb + f * 16 * 64 + sub * 32, 64, 0, lane);
                o[f] = wmma_bf16(pf, vf, o[f]);
            }
        }

        // drain this wave's async copies, then make the new buffer visible
        copy_wait();
        __syncthreads();
    }

    // epilogue: normalize, scatter to ctx[b][s][h*64+d] (bf16 for output GEMM)
    const int b = bh >> 4, h = bh & 15;
#pragma unroll
    for (int r = 0; r < 8; ++r) {
        const float inv = 1.0f / lrow[r];
        const int s = qt * 16 + rowbase + r;
        __bf16* cp = Ctx + ((size_t)(b * S_ + s)) * H_ + h * HD_;
#pragma unroll
        for (int f = 0; f < 4; ++f) cp[f * 16 + ncol] = (__bf16)(o[f][r] * inv);
    }
}

// ---------------------------------------------------------------------------
// Stage 3: output projection  out = ctx @ Wo^T + bo  (fp32 out, 64x32 blocking)
// ---------------------------------------------------------------------------
__global__ void proj_gemm_kernel(const __bf16* __restrict__ Ctxb,
                                 const __bf16* __restrict__ Wob,
                                 const float* __restrict__ bo,
                                 float* __restrict__ out) {
    const int lane = threadIdx.x & 31;
    const int tile = blockIdx.x * (blockDim.x >> 5) + (threadIdx.x >> 5);
    const int ng = tile % 32;               // 32 N-groups of 32 cols
    const int mg = tile / 32;               // 64 M-groups of 64 rows
    const int n0 = ng * 32;

    v8f acc[4][2];
#pragma unroll
    for (int i = 0; i < 4; ++i)
#pragma unroll
        for (int j = 0; j < 2; ++j) acc[i][j] = v8f{0.f,0.f,0.f,0.f,0.f,0.f,0.f,0.f};

    const __bf16* Abase = Ctxb + (size_t)mg * 64 * H_;
    const __bf16* Bbase = Wob + (size_t)n0 * H_;
    for (int kb = 0; kb < H_; kb += 32) {
        v16bf a0 = load_frag_a(Abase + 0 * 16 * H_, H_, kb, lane);
        v16bf a1 = load_frag_a(Abase + 1 * 16 * H_, H_, kb, lane);
        v16bf a2 = load_frag_a(Abase + 2 * 16 * H_, H_, kb, lane);
        v16bf a3 = load_frag_a(Abase + 3 * 16 * H_, H_, kb, lane);
        v16bf b0 = load_frag_b(Bbase + 0 * 16 * H_, H_, kb, lane);
        v16bf b1 = load_frag_b(Bbase + 1 * 16 * H_, H_, kb, lane);
        acc[0][0] = wmma_bf16(a0, b0, acc[0][0]);
        acc[0][1] = wmma_bf16(a0, b1, acc[0][1]);
        acc[1][0] = wmma_bf16(a1, b0, acc[1][0]);
        acc[1][1] = wmma_bf16(a1, b1, acc[1][1]);
        acc[2][0] = wmma_bf16(a2, b0, acc[2][0]);
        acc[2][1] = wmma_bf16(a2, b1, acc[2][1]);
        acc[3][0] = wmma_bf16(a3, b0, acc[3][0]);
        acc[3][1] = wmma_bf16(a3, b1, acc[3][1]);
    }

    const int rowbase = ((lane >> 4) & 1) * 8;
#pragma unroll
    for (int j = 0; j < 2; ++j) {
        const int n = n0 + j * 16 + (lane & 15);
        const float bias = bo[n];
#pragma unroll
        for (int i = 0; i < 4; ++i)
#pragma unroll
            for (int r = 0; r < 8; ++r) {
                int m = mg * 64 + i * 16 + rowbase + r;
                out[(size_t)m * H_ + n] = acc[i][j][r] + bias;
            }
    }
}

// ---------------------------------------------------------------------------
extern "C" void kernel_launch(void* const* d_in, const int* in_sizes, int n_in,
                              void* d_out, int out_size, void* d_ws, size_t ws_size,
                              hipStream_t stream) {
    (void)in_sizes; (void)n_in; (void)out_size; (void)ws_size;

    const float* X  = (const float*)d_in[0];
    const float* Wq = (const float*)d_in[1];
    const float* bq = (const float*)d_in[2];
    const float* Wk = (const float*)d_in[3];
    const float* bk = (const float*)d_in[4];
    const float* Wv = (const float*)d_in[5];
    const float* bv = (const float*)d_in[6];
    const float* Wo = (const float*)d_in[7];
    const float* bo = (const float*)d_in[8];
    float* out = (float*)d_out;

    // workspace layout (bf16 elements), total ~48 MB
    __bf16* base = (__bf16*)d_ws;
    const size_t XN = (size_t)M_ * H_;       // 4 Mi
    const size_t WN = (size_t)H_ * H_;       // 1 Mi
    __bf16* Xb  = base;
    __bf16* Wqb = Xb + XN;
    __bf16* Wkb = Wqb + WN;
    __bf16* Wvb = Wkb + WN;
    __bf16* Wob = Wvb + WN;
    __bf16* Qh  = Wob + WN;
    __bf16* Kh  = Qh + XN;
    __bf16* Vt  = Kh + XN;
    __bf16* Ctx = Vt + XN;

    // Stage 0: fp32 -> bf16
    f2bf4_kernel<<<(int)(XN / 4 / 256), 256, 0, stream>>>(X, Xb, (int)(XN / 4));
    f2bf4_kernel<<<(int)(WN / 4 / 256), 256, 0, stream>>>(Wq, Wqb, (int)(WN / 4));
    f2bf4_kernel<<<(int)(WN / 4 / 256), 256, 0, stream>>>(Wk, Wkb, (int)(WN / 4));
    f2bf4_kernel<<<(int)(WN / 4 / 256), 256, 0, stream>>>(Wv, Wvb, (int)(WN / 4));
    f2bf4_kernel<<<(int)(WN / 4 / 256), 256, 0, stream>>>(Wo, Wob, (int)(WN / 4));

    // Stage 1: QKV projection — 64 M-groups * 96 N-groups = 6144 waves / 8 per block
    qkv_gemm_kernel<<<768, 256, 0, stream>>>(Xb, Wqb, Wkb, Wvb, bq, bk, bv, Qh, Kh, Vt);

    // Stage 2: flash attention — 32 bh * 16 q-blocks = 512 blocks of 8 waves
    attn_kernel<<<512, 256, 0, stream>>>(Qh, Kh, Vt, Ctx);

    // Stage 3: output projection — 64 M-groups * 32 N-groups = 2048 waves / 8 per block
    proj_gemm_kernel<<<256, 256, 0, stream>>>(Ctx, Wob, bo, out);
}
